// OptimizedAdaptiveVQVAE_20315195310708
// MI455X (gfx1250) — compile-verified, matrix operand order checked by reading, not run on hardware
//
#include <hip/hip_runtime.h>
#include <hip/hip_bf16.h>
#include <math.h>

// ---------------------------------------------------------------------------
// VQ-VAE quantization for MI455X (gfx1250, wave32, WMMA + TDM).
// Scores X·E^T via bf16 WMMA with 3-term hi/lo split (near-fp32 accuracy),
// codebook staged to LDS by the Tensor Data Mover, non-temporal stores for
// the 512MB one-hot stream that bounds performance (~27us at 23.3 TB/s).
// ---------------------------------------------------------------------------

typedef __attribute__((ext_vector_type(16))) __bf16        v16bf;
typedef __attribute__((ext_vector_type(8)))  float         v8f;
typedef __attribute__((ext_vector_type(2)))  float         f2v;
typedef __attribute__((ext_vector_type(4)))  unsigned int  v4u;
typedef __attribute__((ext_vector_type(8)))  int           v8i;
typedef __attribute__((ext_vector_type(4)))  int           v4i;

static constexpr int    VQ_N = 262144;   // 64*4096 rows
static constexpr int    VQ_D = 64;
static constexpr int    VQ_K = 512;
static constexpr size_t OFF_Q    = 1;                             // quantized_st
static constexpr size_t OFF_PERP = 1 + (size_t)VQ_N * VQ_D;       // 16777217
static constexpr size_t OFF_ENC  = OFF_PERP + 1;                  // 16777218
static constexpr size_t OFF_IDX  = OFF_ENC + (size_t)VQ_N * VQ_K; // 150994946

// workspace byte offsets (ehi+elo contiguous: one 128KB TDM transfer)
static constexpr size_t WS_EHI  = 0;        // bf16[512*64]  64KB
static constexpr size_t WS_ELO  = 65536;    // bf16[512*64]  64KB
static constexpr size_t WS_EN   = 131072;   // f32[512]       2KB
static constexpr size_t WS_CNT  = 133120;   // u32[512]       2KB
static constexpr size_t WS_LOSS = 135168;   // f32[1]
// LDS: ehi @0 (64KB) | elo @65536 (64KB) | en @131072 (2KB) | idx @133120 (512B)
static constexpr size_t SMEM_BYTES = 133632;

#if __has_builtin(__builtin_amdgcn_tensor_load_to_lds) && \
    __has_builtin(__builtin_amdgcn_s_wait_tensorcnt)
#define VQ_USE_TDM 1
#if __has_include(<hip/amd_detail/amd_gfx1250_TDM.h>)
#define VQ_TDM_6ARG 1
#endif
#endif

union Frag { v16bf v; uint4 q[2]; };

// ---- pre-pass: split embedding into bf16 hi/lo, norms, zero accumulators ---
__global__ void vq_prep(const float* __restrict__ emb,
                        __bf16* __restrict__ ehi, __bf16* __restrict__ elo,
                        float* __restrict__ en, unsigned* __restrict__ counts,
                        float* __restrict__ lossAcc)
{
    int k = blockIdx.x * blockDim.x + threadIdx.x;  // 2 blocks x 256
    if (k < VQ_K) {
        float s = 0.f;
        for (int d = 0; d < VQ_D; ++d) {
            float f = emb[k * VQ_D + d];
            __bf16 h = (__bf16)f;
            ehi[k * VQ_D + d] = h;
            elo[k * VQ_D + d] = (__bf16)(f - (float)h);
            s += f * f;
        }
        en[k] = s;
        counts[k] = 0u;
    }
    if (k == 0) *lossAcc = 0.f;
}

// load one tile's B fragments (hi/lo, both k-steps) from LDS
__device__ __forceinline__ void loadB(Frag* dst, const __bf16* s_ehi,
                                      const __bf16* s_elo, int n0, int hh)
{
    const __bf16* eh = s_ehi + n0 * VQ_D;
    const __bf16* el = s_elo + n0 * VQ_D;
#pragma unroll
    for (int s = 0; s < 2; ++s) {
        const int b2 = 32 * s + 8 * hh;
        dst[2 * s + 0].q[0] = *(const uint4*)(eh + b2);
        dst[2 * s + 0].q[1] = *(const uint4*)(eh + b2 + 16);
        dst[2 * s + 1].q[0] = *(const uint4*)(el + b2);
        dst[2 * s + 1].q[1] = *(const uint4*)(el + b2 + 16);
    }
}

// ---- main kernel: 1 wave = 16 rows; 192 bf16 WMMAs per strip --------------
__global__ __launch_bounds__(256)
void vq_main(const float* __restrict__ inp, const float* __restrict__ emb,
             const unsigned char* __restrict__ mask,
             const __bf16* __restrict__ gehi, const __bf16* __restrict__ gelo,
             const float* __restrict__ gen,
             float* __restrict__ out, unsigned* __restrict__ counts,
             float* __restrict__ lossAcc)
{
    extern __shared__ unsigned char smem[];
    __bf16* s_ehi = (__bf16*)smem;                 // 64KB
    __bf16* s_elo = (__bf16*)(smem + 65536);       // 64KB
    float*  s_en  = (float*)(smem + 131072);       //  2KB (mask pre-folded)
    int*    s_idx = (int*)(smem + 133120);         // 8 waves * 16

    const int tid = threadIdx.x;

#if VQ_USE_TDM
    // --- Tensor Data Mover: DMA the 128KB bf16 hi/lo codebook into LDS ---
    if (tid < 32) {   // one wave issues; TENSORcnt fences before the barrier
        unsigned long long ga = (unsigned long long)(const void*)gehi;
        // D# group0: count=1 | lds_addr=0 | global_addr[56:0] | type=2
        v4u g0 = { 1u, 0u,
                   (unsigned)(ga & 0xffffffffu),
                   (unsigned)((ga >> 32) & 0x1ffffffu) | 0x80000000u };
        // D# group1: data_size=4B; tensor_dim0=tile_dim0=32768 dwords;
        //            tensor_dim1=tile_dim1=1; strides=32768
        v8i g1 = { (int)0x00020000u,   // workgroup_mask=0, data_size=2 (4B)
                   (int)0x80000000u,   // tensor_dim0[15:0]=0x8000 << 16
                   (int)0x00010000u,   // tensor_dim0[31:16]=0, tensor_dim1=1
                   (int)0x80000000u,   // tensor_dim1 hi=0, tile_dim0=0x8000
                   (int)0x00000001u,   // tile_dim1=1, tile_dim2=0
                   (int)0x00008000u,   // tensor_dim0_stride lo = 32768
                   (int)0x80000000u,   // stride0 hi=0, tensor_dim1_stride lo=0x8000
                   0 };                // tensor_dim1_stride hi = 0
        v4i z4 = { 0, 0, 0, 0 };
#if VQ_TDM_6ARG
        v8i z8 = { 0, 0, 0, 0, 0, 0, 0, 0 };
        __builtin_amdgcn_tensor_load_to_lds(g0, g1, z4, z4, z8, 0);
#else
        __builtin_amdgcn_tensor_load_to_lds(g0, g1, z4, z4, 0);
#endif
    }
#else
    {   // fallback: cooperative copy of the hi/lo codebook
        const unsigned* g1 = (const unsigned*)gehi; unsigned* l1 = (unsigned*)s_ehi;
        for (int i = tid; i < 16384; i += 256) l1[i] = g1[i];
        const unsigned* g2 = (const unsigned*)gelo; unsigned* l2 = (unsigned*)s_elo;
        for (int i = tid; i < 16384; i += 256) l2[i] = g2[i];
    }
#endif
    // norms with the active-mask folded in: inactive codes become +huge
    for (int i = tid; i < 512; i += 256)
        s_en[i] = mask[i] ? gen[i] : 3.0e38f;

    const int wave = tid >> 5, lane = tid & 31;
    const int m  = lane & 15;        // row (A) / col (B/C) within 16
    const int hh = lane >> 4;        // half-wave selects k-subblock
    const int row0 = blockIdx.x * 128 + wave * 16;

    // ---- A fragments (overlapped with the TDM transfer) ----
    const float* xrow = inp + (size_t)(row0 + m) * VQ_D;
    v16bf ahi[2], alo[2];
    float part = 0.f;
#pragma unroll
    for (int s = 0; s < 2; ++s) {
        const int b2 = 32 * s + 8 * hh;   // k-chunks: [b2,b2+8) and [b2+16,b2+24)
        float fe[16];
        *(float4*)(fe + 0)  = *(const float4*)(xrow + b2 + 0);
        *(float4*)(fe + 4)  = *(const float4*)(xrow + b2 + 4);
        *(float4*)(fe + 8)  = *(const float4*)(xrow + b2 + 16);
        *(float4*)(fe + 12) = *(const float4*)(xrow + b2 + 20);
#pragma unroll
        for (int e = 0; e < 16; ++e) {
            float f = fe[e];
            __bf16 h = (__bf16)f;
            ahi[s][e] = h;
            alo[s][e] = (__bf16)(f - (float)h);
            part += f * f;
        }
    }
    float xn = part + __shfl_xor(part, 16);    // exact ||x||^2 per row
    float xnr[8];
#pragma unroll
    for (int r = 0; r < 8; ++r) xnr[r] = __shfl(xn, r + 8 * hh);

#if VQ_USE_TDM
    if (tid < 32) __builtin_amdgcn_s_wait_tensorcnt(0);
#endif
    __syncthreads();

    float bestd[8]; int besti[8];
#pragma unroll
    for (int r = 0; r < 8; ++r) { bestd[r] = 3.4e38f; besti[r] = 0; }

    // ---- 32 code tiles, 2-deep ping-pong on B fragments ----
    Frag b[2][4];
    loadB(b[0], s_ehi, s_elo, m, hh);
#pragma unroll 2
    for (int t = 0; t < 32; ++t) {
        const int cur = t & 1, nxt = cur ^ 1;
        if (t < 31) loadB(b[nxt], s_ehi, s_elo, (t + 1) * 16 + m, hh);
        const int n0 = t * 16 + m;          // this lane's code column
        v8f acc = {};
#pragma unroll
        for (int s = 0; s < 2; ++s) {
            acc = __builtin_amdgcn_wmma_f32_16x16x32_bf16(false, ahi[s], false,
                      b[cur][2 * s + 0].v, (short)0, acc, false, false);
            acc = __builtin_amdgcn_wmma_f32_16x16x32_bf16(false, ahi[s], false,
                      b[cur][2 * s + 1].v, (short)0, acc, false, false);
            acc = __builtin_amdgcn_wmma_f32_16x16x32_bf16(false, alo[s], false,
                      b[cur][2 * s + 0].v, (short)0, acc, false, false);
        }
        const float en = s_en[n0];          // +huge for inactive codes
#pragma unroll
        for (int r = 0; r < 8; ++r) {
            float d = xnr[r] + en - 2.f * acc[r];
            if (d < bestd[r]) { bestd[r] = d; besti[r] = n0; }  // earlier tile = lower idx
        }
    }

    // ---- argmin across the 16 lanes that share each row (tie -> lower idx) ----
#pragma unroll
    for (int r = 0; r < 8; ++r) {
#pragma unroll
        for (int off = 1; off < 16; off <<= 1) {
            float od = __shfl_xor(bestd[r], off);
            int   oi = __shfl_xor(besti[r], off);
            if (od < bestd[r] || (od == bestd[r] && oi < besti[r])) {
                bestd[r] = od; besti[r] = oi;
            }
        }
    }
    if (m == 0) {
#pragma unroll
        for (int r = 0; r < 8; ++r) s_idx[wave * 16 + r + 8 * hh] = besti[r];
    }
    __syncthreads();   // uniform across block; broadcasts indices wave-wide

    // ---- write phase: quantized (exact fp32 gather), fused one-hot, loss ----
    // All output streams are written once and never re-read -> non-temporal,
    // keeping the codebook/inputs resident in the 192MB L2.
    float* outQ = out + OFF_Q;
    float* outE = out + OFF_ENC;
    float lacc = 0.f;
    for (int mm = 0; mm < 16; ++mm) {
        const int row  = row0 + mm;
        const int code = s_idx[wave * 16 + mm];
        {   // quantized_st + squared-error, 2 elements per lane
            const int d0 = lane * 2;
            f2v q = *(const f2v*)(emb + (size_t)code * VQ_D + d0);
            f2v x = *(const f2v*)(inp + (size_t)row  * VQ_D + d0);
            __builtin_nontemporal_store(q.x, outQ + (size_t)row * VQ_D + d0 + 0);
            __builtin_nontemporal_store(q.y, outQ + (size_t)row * VQ_D + d0 + 1);
            float dx = q.x - x.x, dy = q.y - x.y;
            lacc += dx * dx + dy * dy;
        }
        // one-hot row of 512 floats; f2v stores keep 8B alignment (OFF_ENC even)
        float* erow = outE + (size_t)row * VQ_K;
#pragma unroll
        for (int i = 0; i < 8; ++i) {
            const int j = i * 64 + lane * 2;
            f2v v;
            v.x = (j     == code) ? 1.f : 0.f;
            v.y = (j + 1 == code) ? 1.f : 0.f;
            __builtin_nontemporal_store(v, (f2v*)(erow + j));
        }
    }
    if (lane < 16) {   // indices (as float values) + histogram
        const int code = s_idx[wave * 16 + lane];
        __builtin_nontemporal_store((float)code, out + OFF_IDX + row0 + lane);
        atomicAdd(&counts[code], 1u);
    }
#pragma unroll
    for (int off = 16; off; off >>= 1) lacc += __shfl_xor(lacc, off);
    if (lane == 0) atomicAdd(lossAcc, lacc);
}

// ---- finalize: perplexity + vq_loss scalars --------------------------------
__global__ void vq_finalize(const unsigned* __restrict__ counts,
                            const float* __restrict__ lossAcc,
                            float* __restrict__ out)
{
    __shared__ float red[256];
    const int tid = threadIdx.x;
    float s = 0.f;
    for (int k = tid; k < VQ_K; k += 256) {
        float p = (float)counts[k] * (1.f / (float)VQ_N);
        s += p * logf(p + 1e-10f);
    }
    red[tid] = s; __syncthreads();
    for (int o = 128; o; o >>= 1) { if (tid < o) red[tid] += red[tid + o]; __syncthreads(); }
    if (tid == 0) {
        out[OFF_PERP] = expf(-red[0]);
        float msq = lossAcc[0] / (float)((size_t)VQ_N * VQ_D);
        out[0] = 1.25f * msq;   // q_latent + 0.25 * e_latent (numerically equal)
    }
}

extern "C" void kernel_launch(void* const* d_in, const int* in_sizes, int n_in,
                              void* d_out, int out_size, void* d_ws, size_t ws_size,
                              hipStream_t stream)
{
    (void)in_sizes; (void)n_in; (void)out_size; (void)ws_size;
    const float*         inp  = (const float*)d_in[0];
    const float*         emb  = (const float*)d_in[1];
    const unsigned char* mask = (const unsigned char*)d_in[2];
    float* out = (float*)d_out;

    char* ws = (char*)d_ws;
    __bf16*   ehi    = (__bf16*)(ws + WS_EHI);
    __bf16*   elo    = (__bf16*)(ws + WS_ELO);
    float*    en     = (float*)(ws + WS_EN);
    unsigned* counts = (unsigned*)(ws + WS_CNT);
    float*    loss   = (float*)(ws + WS_LOSS);

    vq_prep<<<2, 256, 0, stream>>>(emb, ehi, elo, en, counts, loss);
    vq_main<<<VQ_N / 128, 256, SMEM_BYTES, stream>>>(inp, emb, mask, ehi, elo, en,
                                                     out, counts, loss);
    vq_finalize<<<1, 256, 0, stream>>>(counts, loss, out);
}
